// AttnLayer_32495722561882
// MI455X (gfx1250) — matrix operand order repaired and checked by reading, not compiled, not run on hardware
//
#include <hip/hip_runtime.h>
#include <math.h>

typedef __bf16 bf16;
typedef __attribute__((ext_vector_type(16))) __bf16 v16bf;
typedef __attribute__((ext_vector_type(8)))  __bf16 v8bf;
typedef __attribute__((ext_vector_type(8)))  float  v8f;

#define SMEM_BYTES 114688

// ---- helpers -------------------------------------------------------------

// A-matrix (16x32 bf16) fragment from a row-major buffer.
// lane l holds row (tile*16 + l%16); k = e + (l>=16?16:0) within a
// 32-wide k-chunk  -> two contiguous 16B chunks at +0 and +8 elements.
// p must already point at  rowbase + kchunk_base + (lane>>4)*16.
__device__ inline v16bf a_frag(const bf16* p) {
  v8bf c0 = *(const v8bf*)p;
  v8bf c1 = *(const v8bf*)(p + 8);
  v16bf r;
#pragma unroll
  for (int i = 0; i < 8; ++i) { r[i] = c0[i]; r[i + 8] = c1[i]; }
  return r;
}

__device__ inline v8f wmma_bf16(v16bf a, v16bf b, v8f c) {
  return __builtin_amdgcn_wmma_f32_16x16x32_bf16(false, a, false, b,
                                                 (short)0, c, false, false);
}

__device__ inline float softmax_at(const float* w, int n, int idx) {
  float mx = w[0];
  for (int i = 1; i < n; ++i) mx = fmaxf(mx, w[i]);
  float s = 0.f;
  for (int i = 0; i < n; ++i) s += expf(w[i] - mx);
  return expf(w[idx] - mx) / s;
}

// ---- elementwise kernels -------------------------------------------------

__global__ __launch_bounds__(256) void f32_to_bf16(const float* __restrict__ src,
                                                   bf16* __restrict__ dst, long n) {
  long stride = (long)gridDim.x * 256;
  for (long i = (long)blockIdx.x * 256 + threadIdx.x; i < n; i += stride)
    dst[i] = (bf16)src[i];
}

// Pack all 24 weight matrices (6 sets x {wq,wk,wv,wo}, each 256x256 row-major,
// used as X @ W^T) into WMMA B-fragment order:
//   idx = ((g*16 + nt)*8 + kc)*32*16 + lane*16 + e
//   value = W[n = nt*16 + lane%16][k = kc*32 + (lane/16)*16 + e]
__global__ __launch_bounds__(256) void pack_weights(const float* __restrict__ wqkv,
                                                    const float* __restrict__ wo,
                                                    bf16* __restrict__ wpack) {
  const long total = 24L * 65536;
  long stride = (long)gridDim.x * 256;
  for (long i = (long)blockIdx.x * 256 + threadIdx.x; i < total; i += stride) {
    int g = (int)(i >> 16);
    int r = (int)(i & 65535);
    int e    = r & 15;
    int lane = (r >> 4) & 31;
    int kc   = (r >> 9) & 7;
    int nt   = (r >> 12) & 15;
    int set = g >> 2, m = g & 3;
    int n = nt * 16 + (lane & 15);
    int k = kc * 32 + ((lane >> 4) * 16) + e;
    float v = (m < 3) ? wqkv[((long)set * 768 + m * 256 + n) * 256 + k]
                      : wo[((long)set * 256 + n) * 256 + k];
    wpack[i] = (bf16)v;
  }
}

// out[i] = softmax(wvec)[idx] * x[i]   (residual-term init of d_out)
__global__ __launch_bounds__(256) void scale_into(const float* __restrict__ x,
                                                  float* __restrict__ out,
                                                  const float* __restrict__ wvec,
                                                  int nw, int idx, long n) {
  float alpha = softmax_at(wvec, nw, idx);
  long stride = (long)gridDim.x * 256;
  for (long i = (long)blockIdx.x * 256 + threadIdx.x; i < n; i += stride)
    out[i] = alpha * x[i];
}

// ---- fully fused MHA: one block per batch element ------------------------
// d_out[b] += softmax(wvec)[widx] * MHA(Xq[b], Xkv[b])
__global__ __launch_bounds__(256) void fused_mha(
    const bf16* __restrict__ Xq, long qstride,
    const bf16* __restrict__ Xkv, long kvstride,
    int Sq, int Sk,
    const bf16* __restrict__ wfrag, int set,
    const float* __restrict__ bqkv_set,  // 768 floats: bq|bk|bv
    const float* __restrict__ bo_set,    // 256 floats
    float* __restrict__ outp, long ostride,
    const float* __restrict__ wvec, int nw, int widx) {
  extern __shared__ char smem[];
  bf16*  lq    = (bf16*)(smem + 0);       // [32][256] q input (zero padded)
  bf16*  lkv   = (bf16*)(smem + 16384);   // [32][256] kv input (zero padded)
  bf16*  qh    = (bf16*)(smem + 32768);   // [32][256]
  bf16*  kh    = (bf16*)(smem + 49152);   // [32][256]
  bf16*  vT    = (bf16*)(smem + 65536);   // [256][32] v projection, transposed
  float* sc    = (float*)(smem + 0);      // [8][32][32] overlays lq/lkv
  bf16*  attnP = (bf16*)(smem + 81920);   // [8][32][32] softmax probs, padded 0
  bf16*  aout  = (bf16*)(smem + 98304);   // [32][256] attention output

  const int tid  = threadIdx.x;
  const int wave = tid >> 5;
  const int lane = tid & 31;
  const int lhi  = lane >> 4;   // 0/1
  const int llo  = lane & 15;
  const int b    = blockIdx.x;

  const float alpha = softmax_at(wvec, nw, widx);

  // ---- Phase A: stage inputs into LDS (zero-pad to 32 rows) ----
  for (int i = tid; i < 32 * 256; i += 256) {
    int r = i >> 8, c = i & 255;
    lq[i]  = (r < Sq) ? Xq[(long)b * qstride + r * 256 + c]  : (bf16)0.f;
    lkv[i] = (r < Sk) ? Xkv[(long)b * kvstride + r * 256 + c] : (bf16)0.f;
  }
  __syncthreads();

  // ---- Phase B: Q/K/V projections (96 16x16 tiles, 8 K-steps each) ----
  for (int t = wave; t < 96; t += 8) {
    int m = t / 32;               // 0=q, 1=k, 2=v
    int sub = t - m * 32;
    int mt = sub >> 4, nt = sub & 15;
    const bf16* X  = (m == 0) ? lq : lkv;
    const bf16* wf = wfrag + ((long)(set * 4 + m) * 16 + nt) * 4096;
    const bf16* arow = X + (mt * 16 + llo) * 256 + lhi * 16;
    v8f acc = {};
#pragma unroll
    for (int kc = 0; kc < 8; ++kc) {
      v16bf a  = a_frag(arow + kc * 32);
      v16bf bb = *(const v16bf*)(wf + (kc * 32 + lane) * 16);
      acc = wmma_bf16(a, bb, acc);
    }
    int n = nt * 16 + llo;
    float bias = bqkv_set[m * 256 + n];
    int rbase = mt * 16 + lhi * 8;
#pragma unroll
    for (int r = 0; r < 8; ++r) {
      float v = acc[r] + bias;
      int rr = rbase + r;
      if (m == 0)      qh[rr * 256 + n] = (bf16)v;
      else if (m == 1) kh[rr * 256 + n] = (bf16)v;
      else             vT[n * 32 + rr]  = (bf16)v;   // transposed store
    }
  }
  __syncthreads();

  // ---- Phase C: scores = qh @ kh^T / sqrt(hd), wave == head ----
  {
    int h = wave;
#pragma unroll
    for (int c4 = 0; c4 < 4; ++c4) {
      int mt = c4 >> 1, st = c4 & 1;
      v16bf a  = a_frag(qh + (mt * 16 + llo) * 256 + h * 32 + lhi * 16);
      v16bf bb = *(const v16bf*)(kh + (st * 16 + llo) * 256 + h * 32 + lhi * 16);
      v8f acc = {};
      acc = wmma_bf16(a, bb, acc);
      int scol = st * 16 + llo;
      int rbase = mt * 16 + lhi * 8;
#pragma unroll
      for (int r = 0; r < 8; ++r)
        sc[(h * 32 + rbase + r) * 32 + scol] = acc[r] * 0.17677669529663687f;
    }
  }
  __syncthreads();

  // ---- Phase D: row softmax, masked to valid Sk; pad probs with 0 ----
  {
    int h = tid >> 5, q = tid & 31;
    float* srow = sc + (h * 32 + q) * 32;
    float mx = -3.0e38f;
    for (int s = 0; s < Sk; ++s) mx = fmaxf(mx, srow[s]);
    float sum = 0.f;
    for (int s = 0; s < Sk; ++s) sum += expf(srow[s] - mx);
    float inv = 1.f / sum;
    bf16* ap = attnP + (h * 32 + q) * 32;
    for (int s = 0; s < 32; ++s)
      ap[s] = (s < Sk) ? (bf16)(expf(srow[s] - mx) * inv) : (bf16)0.f;
  }
  __syncthreads();

  // ---- Phase E: out_h = attn @ vh, wave == head ----
  {
    int h = wave;
#pragma unroll
    for (int c4 = 0; c4 < 4; ++c4) {
      int mt = c4 >> 1, nt2 = c4 & 1;
      v16bf a = a_frag(attnP + (h * 32 + mt * 16 + llo) * 32 + lhi * 16);
      int dcol = h * 32 + nt2 * 16 + llo;
      v16bf bb = *(const v16bf*)(vT + dcol * 32 + lhi * 16);
      v8f acc = {};
      acc = wmma_bf16(a, bb, acc);
      int rbase = mt * 16 + lhi * 8;
#pragma unroll
      for (int r = 0; r < 8; ++r)
        aout[(rbase + r) * 256 + dcol] = (bf16)acc[r];
    }
  }
  __syncthreads();

  // ---- Phase F: output projection + scaled accumulate into d_out ----
  for (int t = wave; t < 32; t += 8) {
    int mt = t >> 4, nt = t & 15;
    const bf16* wf = wfrag + ((long)(set * 4 + 3) * 16 + nt) * 4096;
    const bf16* arow = aout + (mt * 16 + llo) * 256 + lhi * 16;
    v8f acc = {};
#pragma unroll
    for (int kc = 0; kc < 8; ++kc) {
      v16bf a  = a_frag(arow + kc * 32);
      v16bf bb = *(const v16bf*)(wf + (kc * 32 + lane) * 16);
      acc = wmma_bf16(a, bb, acc);
    }
    int n = nt * 16 + llo;
    float bias = bo_set[n];
    int rbase = mt * 16 + lhi * 8;
#pragma unroll
    for (int r = 0; r < 8; ++r) {
      int rr = rbase + r;
      if (rr < Sq) {
        float* op = outp + (long)b * ostride + rr * 256 + n;
        *op += alpha * (acc[r] + bias);
      }
    }
  }
}

// ---- host launcher -------------------------------------------------------

extern "C" void kernel_launch(void* const* d_in, const int* in_sizes, int n_in,
                              void* d_out, int out_size, void* d_ws, size_t ws_size,
                              hipStream_t stream) {
  const float* leaf   = (const float*)d_in[0];
  const float* inner  = (const float*)d_in[1];
  const float* root   = (const float*)d_in[2];
  const float* wqkv   = (const float*)d_in[3];
  const float* bqkv   = (const float*)d_in[4];
  const float* wo     = (const float*)d_in[5];
  const float* bo     = (const float*)d_in[6];
  const float* wleaf  = (const float*)d_in[7];
  const float* winner = (const float*)d_in[8];
  const float* wroot  = (const float*)d_in[9];

  const long NLEAF = 33554432L, NINNER = 1048576L, NROOT = 131072L;

  char* ws = (char*)d_ws;
  bf16* leaf_bf  = (bf16*)ws;                   // 67,108,864 B
  bf16* inner_bf = (bf16*)(ws + 67108864);      //  2,097,152 B
  bf16* root_bf  = (bf16*)(ws + 69206016);      //    262,144 B
  bf16* wpack    = (bf16*)(ws + 69468160);      //  3,145,728 B  (24 x 256x256)

  float* out_leaf  = (float*)d_out;
  float* out_inner = out_leaf + NLEAF;
  float* out_root  = out_inner + NINNER;

  (void)hipFuncSetAttribute((const void*)fused_mha,
                            hipFuncAttributeMaxDynamicSharedMemorySize, SMEM_BYTES);

  // bf16 conversions + weight fragment packing
  f32_to_bf16<<<2048, 256, 0, stream>>>(leaf, leaf_bf, NLEAF);
  f32_to_bf16<<<512, 256, 0, stream>>>(inner, inner_bf, NINNER);
  f32_to_bf16<<<128, 256, 0, stream>>>(root, root_bf, NROOT);
  pack_weights<<<1024, 256, 0, stream>>>(wqkv, wo, wpack);

  // init d_out with the residual terms
  scale_into<<<2048, 256, 0, stream>>>(leaf, out_leaf, wleaf, 3, 2, NLEAF);
  scale_into<<<512, 256, 0, stream>>>(inner, out_inner, winner, 4, 3, NINNER);
  scale_into<<<128, 256, 0, stream>>>(root, out_root, wroot, 3, 2, NROOT);

  // 7 MHA instances, each accumulating alpha * mha(...) into its region.
  // leaf_p: q=leaf, kv=inner_flat (Sk=1), set 0
  fused_mha<<<4096, 256, SMEM_BYTES, stream>>>(leaf_bf, 8192, inner_bf, 256, 32, 1,
      wpack, 0, bqkv + 0 * 768, bo + 0 * 256, out_leaf, 8192, wleaf, 3, 0);
  // leaf_s: self-attn on leaf, set 1
  fused_mha<<<4096, 256, SMEM_BYTES, stream>>>(leaf_bf, 8192, leaf_bf, 8192, 32, 32,
      wpack, 1, bqkv + 1 * 768, bo + 1 * 256, out_leaf, 8192, wleaf, 3, 1);
  // inner_p: q=inner, kv=root (Sk=4), set 2
  fused_mha<<<128, 256, SMEM_BYTES, stream>>>(inner_bf, 8192, root_bf, 1024, 32, 4,
      wpack, 2, bqkv + 2 * 768, bo + 2 * 256, out_inner, 8192, winner, 4, 0);
  // inner_s: self-attn on inner, set 3
  fused_mha<<<128, 256, SMEM_BYTES, stream>>>(inner_bf, 8192, inner_bf, 8192, 32, 32,
      wpack, 3, bqkv + 3 * 768, bo + 3 * 256, out_inner, 8192, winner, 4, 1);
  // inner_c: q=inner_flat (Sq=1), kv=leaf, set 4
  fused_mha<<<4096, 256, SMEM_BYTES, stream>>>(inner_bf, 256, leaf_bf, 8192, 1, 32,
      wpack, 4, bqkv + 4 * 768, bo + 4 * 256, out_inner, 256, winner, 4, 2);
  // root_s: self-attn on root (Sq=Sk=4), set 5
  fused_mha<<<128, 256, SMEM_BYTES, stream>>>(root_bf, 1024, root_bf, 1024, 4, 4,
      wpack, 5, bqkv + 5 * 768, bo + 5 * 256, out_root, 1024, wroot, 3, 0);
  // root_c: q=root (Sq=4), kv=inner, set 5 (same weights as root_s)
  fused_mha<<<128, 256, SMEM_BYTES, stream>>>(root_bf, 1024, inner_bf, 8192, 4, 32,
      wpack, 5, bqkv + 5 * 768, bo + 5 * 256, out_root, 1024, wroot, 3, 1);
}
